// PyTorchAttentionLayer_v01_20985210208303
// MI455X (gfx1250) — matrix operand order repaired and checked by reading, not compiled, not run on hardware
//
#include <hip/hip_runtime.h>
#include <math.h>
#include <stdint.h>

// Problem constants (match reference)
#define Bd    2
#define Sd    2048
#define Ed    512
#define Hd    8
#define EDGEd 16
#define Dd    64

typedef _Float16 v16h __attribute__((ext_vector_type(16)));
typedef _Float16 v8h  __attribute__((ext_vector_type(8)));
typedef _Float16 v4h  __attribute__((ext_vector_type(4)));
typedef float    v8f  __attribute__((ext_vector_type(8)));

static __device__ __forceinline__ v8f wmma_f16(v16h a, v16h b, v8f c) {
  // 8 args: (neg_a, A, neg_b, B, c_mod, C, reuse_a, reuse_b)
  return __builtin_amdgcn_wmma_f32_16x16x32_f16(false, a, false, b, (short)0, c,
                                                false, false);
}

static __device__ __forceinline__ v16h cat8(v8h lo, v8h hi) {
  return __builtin_shufflevector(lo, hi, 0, 1, 2, 3, 4, 5, 6, 7, 8, 9, 10, 11,
                                 12, 13, 14, 15);
}

// ---------------------------------------------------------------------------
// Kernel 0: one-pass f32 -> f16 conversion (x inputs and weight matrices).
// Removes per-tile re-conversion from the GEMM inner loops.
// ---------------------------------------------------------------------------
__global__ __launch_bounds__(256) void cvt_f16_kernel(
    const float* __restrict__ src, _Float16* __restrict__ dst, int n4) {
  int i = blockIdx.x * 256 + threadIdx.x;
  const int stride = gridDim.x * 256;
  for (; i < n4; i += stride) {
    const float4 v = ((const float4*)src)[i];
    v4h h = {(_Float16)v.x, (_Float16)v.y, (_Float16)v.z, (_Float16)v.w};
    ((v4h*)dst)[i] = h;
  }
}

// ---------------------------------------------------------------------------
// Kernel 1: packed QKV projection (f16 in, f32 WMMA accumulate).
//   qh, kh: [B,H,S,D] f16   vT: [B,H,D,S] f16 (so PV B-frags load contiguously)
// ---------------------------------------------------------------------------
__global__ __launch_bounds__(256) void qkv_proj_kernel(
    const _Float16* __restrict__ xq, const _Float16* __restrict__ xk,
    const _Float16* __restrict__ xv, const _Float16* __restrict__ W16,
    const float* __restrict__ bvec, _Float16* __restrict__ qh,
    _Float16* __restrict__ kh, _Float16* __restrict__ vT) {
  const int lane = threadIdx.x & 31;
  const int wave = threadIdx.x >> 5;
  const int tile = blockIdx.x * 8 + wave;
  const int Nt = Ed / 16;                        // 32
  const int tilesPerProj = (Bd * Sd / 16) * Nt;  // 8192
  const int proj = tile / tilesPerProj;
  const int rem = tile - proj * tilesPerProj;
  const int mt = rem / Nt, nt = rem - mt * Nt;
  const int m0 = mt * 16, n0 = nt * 16;
  const int r = lane & 15, g = lane >> 4;

  const _Float16* X = (proj == 0) ? xq : (proj == 1) ? xk : xv;
  const _Float16* Wp = W16 + (size_t)proj * Ed * Ed;

  const _Float16* arow = X + (size_t)(m0 + r) * Ed + g * 16;
  const _Float16* brow = Wp + (size_t)(n0 + r) * Ed + g * 16;

  v8f acc;
#pragma unroll
  for (int i = 0; i < 8; ++i) acc[i] = 0.0f;

#pragma unroll 4
  for (int kk = 0; kk < Ed; kk += 32) {
    v16h af = cat8(*(const v8h*)(arow + kk), *(const v8h*)(arow + kk + 8));
    v16h bf = *(const v16h*)(brow + kk);
    acc = wmma_f16(af, bf, acc);
  }

  const int ncol = n0 + r;
  const float bn = bvec[proj * Ed + ncol];
  const int h = ncol >> 6, d = ncol & 63;
#pragma unroll
  for (int rr = 0; rr < 8; ++rr) {
    const int m = m0 + rr + 8 * g;
    const int bb = m >> 11, s = m & (Sd - 1);
    const _Float16 val = (_Float16)(acc[rr] + bn);
    if (proj == 0)
      qh[(((size_t)bb * Hd + h) * Sd + s) * Dd + d] = val;
    else if (proj == 1)
      kh[(((size_t)bb * Hd + h) * Sd + s) * Dd + d] = val;
    else
      vT[(((size_t)bb * Hd + h) * Dd + d) * Sd + s] = val;
  }
}

// ---------------------------------------------------------------------------
// Kernel 2: flash attention with fused edge bias.
// Block = (b, 16-row q-tile); 8 waves = 8 heads share the p tile (p is read
// exactly once from HBM = the 537 MB roofline floor). p tiles stream through
// LDS via double-buffered global_load_async_to_lds_b128 (ASYNCcnt) so HBM
// latency overlaps WMMA + softmax of the previous tile.
// ---------------------------------------------------------------------------
__global__ __launch_bounds__(256) void attn_kernel(
    const _Float16* __restrict__ qh, const _Float16* __restrict__ kh,
    const _Float16* __restrict__ vT, const float* __restrict__ p,
    const float* __restrict__ edge_w, const float* __restrict__ edge_b,
    _Float16* __restrict__ ctx) {
  __shared__ float pt[2][16][32][EDGEd];  // 64 KB: double-buffered p tiles
  __shared__ _Float16 ps[8][16][32];      // 8 KB: per-wave P staging (C->A)

  const int lane = threadIdx.x & 31;
  const int h = threadIdx.x >> 5;  // wave id == head
  const int b = blockIdx.x >> 7;   // S/16 = 128 q-tiles per batch
  const int q0 = (blockIdx.x & 127) * 16;
  const int r = lane & 15, g = lane >> 4;

  const float* pbase = p + ((size_t)b * Sd + q0) * Sd * EDGEd;

  // Issue the 8 per-thread async 16B copies for one 16x32x16 p tile.
  auto issue_tile = [&](int k0, int sel) {
#pragma unroll
    for (int j = 0; j < 8; ++j) {
      const int i = threadIdx.x + j * 256;
      const int qi = i >> 7;
      const int ki = (i >> 2) & 31;
      const int e4 = i & 3;
      const float* gsrc = pbase + ((size_t)qi * Sd + k0 + ki) * EDGEd + e4 * 4;
      const unsigned ldsa =
          (unsigned)(uintptr_t)(void*)&pt[sel][qi][ki][e4 * 4];
      asm volatile("global_load_async_to_lds_b128 %0, %1, off" ::"v"(ldsa),
                   "v"(gsrc)
                   : "memory");
    }
  };

  // Q A-fragments for d=0..31 and d=32..63 (live in registers all k-steps)
  v16h qa0, qa1;
  {
    const _Float16* qb = qh + (((size_t)b * Hd + h) * Sd + q0 + r) * Dd + g * 16;
    qa0 = cat8(*(const v8h*)(qb), *(const v8h*)(qb + 8));
    qa1 = cat8(*(const v8h*)(qb + 32), *(const v8h*)(qb + 40));
  }

  float we[EDGEd];
#pragma unroll
  for (int e = 0; e < EDGEd; ++e) we[e] = edge_w[h * EDGEd + e];
  const float eb = edge_b[h];

  v8f o[4];
#pragma unroll
  for (int c = 0; c < 4; ++c)
#pragma unroll
    for (int i = 0; i < 8; ++i) o[c][i] = 0.0f;
  float mrow[8], lrow[8];
#pragma unroll
  for (int i = 0; i < 8; ++i) {
    mrow[i] = -3.0e38f;
    lrow[i] = 0.0f;
  }

  const float scale = 0.125f;  // 1/sqrt(64)
  const _Float16* kbase = kh + (((size_t)b * Hd + h) * Sd) * Dd;
  const _Float16* vbase = vT + (((size_t)b * Hd + h) * Dd) * Sd;

  issue_tile(0, 0);  // prologue: tile 0 in flight

  const int nT = Sd / 32;  // 64
  for (int t = 0; t < nT; ++t) {
    const int sel = t & 1;
    const int k0 = t * 32;
    if (t + 1 < nT) {
      issue_tile(k0 + 32, sel ^ 1);  // overlap next tile's HBM traffic
      // Async loads complete in order: <=8 outstanding => tile t's 8 are done.
      asm volatile("s_wait_asynccnt 0x8" ::: "memory");
    } else {
      asm volatile("s_wait_asynccnt 0x0" ::: "memory");
    }
    __syncthreads();  // tile t visible to all waves

    // Scores: two 16x16 fragments (key cols k0..+15, k0+16..+31)
    v8f sf[2];
#pragma unroll
    for (int f = 0; f < 2; ++f) {
      const _Float16* kb = kbase + (size_t)(k0 + f * 16 + r) * Dd + g * 16;
      v16h b0 = *(const v16h*)(kb);       // contraction d = g*16 + 0..15
      v16h b1 = *(const v16h*)(kb + 32);  // contraction d = 32 + g*16 + 0..15
      v8f c;
#pragma unroll
      for (int i = 0; i < 8; ++i) c[i] = 0.0f;
      c = wmma_f16(qa0, b0, c);
      c = wmma_f16(qa1, b1, c);
      const int kcol = f * 16 + r;
#pragma unroll
      for (int rr = 0; rr < 8; ++rr) {
        const int qi = rr + 8 * g;
        const float* pe = &pt[sel][qi][kcol][0];
        float bias = eb;
#pragma unroll
        for (int e = 0; e < EDGEd; ++e) bias += pe[e] * we[e];
        sf[f][rr] = c[rr] * scale + bias;
      }
    }

    // Online softmax (row reductions across the 16-lane C-layout groups)
#pragma unroll
    for (int rr = 0; rr < 8; ++rr) {
      float mx = fmaxf(sf[0][rr], sf[1][rr]);
#pragma unroll
      for (int msk = 8; msk >= 1; msk >>= 1)
        mx = fmaxf(mx, __shfl_xor(mx, msk, 32));
      const float mnew = fmaxf(mrow[rr], mx);
      const float a = __expf(mrow[rr] - mnew);
      const float p0 = __expf(sf[0][rr] - mnew);
      const float p1 = __expf(sf[1][rr] - mnew);
      sf[0][rr] = p0;
      sf[1][rr] = p1;
      float rs = p0 + p1;
#pragma unroll
      for (int msk = 8; msk >= 1; msk >>= 1) rs += __shfl_xor(rs, msk, 32);
      lrow[rr] = lrow[rr] * a + rs;
      mrow[rr] = mnew;
#pragma unroll
      for (int c = 0; c < 4; ++c) o[c][rr] *= a;
    }

    // Re-layout P: C layout -> LDS -> A layout (16x32 f16), wave-private region
#pragma unroll
    for (int f = 0; f < 2; ++f)
#pragma unroll
      for (int rr = 0; rr < 8; ++rr)
        ps[h][rr + 8 * g][f * 16 + r] = (_Float16)sf[f][rr];
    asm volatile("s_wait_dscnt 0" ::: "memory");  // wave-local LDS RAW fence
    v16h pa;
    {
      const _Float16* pp = &ps[h][r][g * 16];
      pa = cat8(*(const v8h*)(pp), *(const v8h*)(pp + 8));
    }

    // PV: O[16q x 64d] += P[16x32] * V[32x16] per d-chunk (vT contiguous in k)
#pragma unroll
    for (int c = 0; c < 4; ++c) {
      const _Float16* vb = vbase + (size_t)(c * 16 + r) * Sd + k0 + g * 16;
      v16h bv = *(const v16h*)vb;
      o[c] = wmma_f16(pa, bv, o[c]);
    }

    __syncthreads();  // all waves done reading pt[sel] before its reuse (t+2)
  }

  // Normalize and store ctx as [B,S,E] f16 (e = h*64 + c*16 + col)
#pragma unroll
  for (int rr = 0; rr < 8; ++rr) {
    const float inv = 1.0f / lrow[rr];
#pragma unroll
    for (int c = 0; c < 4; ++c) o[c][rr] *= inv;
  }
#pragma unroll
  for (int c = 0; c < 4; ++c)
#pragma unroll
    for (int rr = 0; rr < 8; ++rr)
      ctx[((size_t)b * Sd + q0 + rr + 8 * g) * Ed + h * 64 + c * 16 + r] =
          (_Float16)o[c][rr];
}

// ---------------------------------------------------------------------------
// Kernel 3: output projection. out = ctx @ W^T + b  (f32 output)
// ---------------------------------------------------------------------------
__global__ __launch_bounds__(256) void out_proj_kernel(
    const _Float16* __restrict__ ctx, const _Float16* __restrict__ W16,
    const float* __restrict__ bvec, float* __restrict__ out) {
  const int lane = threadIdx.x & 31;
  const int wave = threadIdx.x >> 5;
  const int tile = blockIdx.x * 8 + wave;
  const int Nt = Ed / 16;  // 32
  const int mt = tile / Nt, nt = tile - mt * Nt;
  const int m0 = mt * 16, n0 = nt * 16;
  const int r = lane & 15, g = lane >> 4;

  const _Float16* arow = ctx + (size_t)(m0 + r) * Ed + g * 16;
  const _Float16* brow = W16 + (size_t)(n0 + r) * Ed + g * 16;

  v8f acc;
#pragma unroll
  for (int i = 0; i < 8; ++i) acc[i] = 0.0f;

#pragma unroll 4
  for (int kk = 0; kk < Ed; kk += 32) {
    v16h af = cat8(*(const v8h*)(arow + kk), *(const v8h*)(arow + kk + 8));
    v16h bf = *(const v16h*)(brow + kk);
    acc = wmma_f16(af, bf, acc);
  }

  const float bn = bvec[n0 + r];
#pragma unroll
  for (int rr = 0; rr < 8; ++rr)
    out[(size_t)(m0 + rr + 8 * g) * Ed + n0 + r] = acc[rr] + bn;
}

// ---------------------------------------------------------------------------
extern "C" void kernel_launch(void* const* d_in, const int* in_sizes, int n_in,
                              void* d_out, int out_size, void* d_ws,
                              size_t ws_size, hipStream_t stream) {
  const float* q = (const float*)d_in[0];
  const float* k = (const float*)d_in[1];
  const float* v = (const float*)d_in[2];
  const float* p = (const float*)d_in[3];
  const float* in_w = (const float*)d_in[4];
  const float* in_b = (const float*)d_in[5];
  const float* out_w = (const float*)d_in[6];
  const float* out_b = (const float*)d_in[7];
  const float* ew = (const float*)d_in[8];
  const float* ebp = (const float*)d_in[9];

  const size_t nTok = (size_t)Bd * Sd * Ed;  // 2,097,152
  _Float16* qh = (_Float16*)d_ws;            // [B,H,S,D]
  _Float16* kh = qh + nTok;                  // [B,H,S,D]
  _Float16* vT = kh + nTok;                  // [B,H,D,S]
  _Float16* ctx = vT + nTok;                 // [B,S,E]
  _Float16* x16q = ctx + nTok;               // f16 copies of inputs/weights
  _Float16* x16k = x16q + nTok;
  _Float16* x16v = x16k + nTok;
  _Float16* w16in = x16v + nTok;                      // 3*E*E
  _Float16* w16out = w16in + (size_t)3 * Ed * Ed;     // E*E
  float* out = (float*)d_out;                         // ws use ~31.5 MB total

  // One-pass f32->f16 of streaming GEMM operands.
  cvt_f16_kernel<<<512, 256, 0, stream>>>(q, x16q, (int)(nTok / 4));
  cvt_f16_kernel<<<512, 256, 0, stream>>>(k, x16k, (int)(nTok / 4));
  cvt_f16_kernel<<<512, 256, 0, stream>>>(v, x16v, (int)(nTok / 4));
  cvt_f16_kernel<<<256, 256, 0, stream>>>(in_w, w16in, 3 * Ed * Ed / 4);
  cvt_f16_kernel<<<128, 256, 0, stream>>>(out_w, w16out, Ed * Ed / 4);

  const int projBlocks = 3 * (Bd * Sd / 16) * (Ed / 16) / 8;  // 3072
  qkv_proj_kernel<<<projBlocks, 256, 0, stream>>>(x16q, x16k, x16v, w16in,
                                                  in_b, qh, kh, vT);

  attn_kernel<<<Bd * (Sd / 16), 256, 0, stream>>>(qh, kh, vT, p, ew, ebp, ctx);

  const int outBlocks = (Bd * Sd / 16) * (Ed / 16) / 8;  // 1024
  out_proj_kernel<<<outBlocks, 256, 0, stream>>>(ctx, w16out, out_b, out);
}
